// HybridRetriever_78323023609925
// MI455X (gfx1250) — matrix operand order repaired and checked by reading, not compiled, not run on hardware
//
#include <hip/hip_runtime.h>
#include <math.h>

typedef __attribute__((ext_vector_type(16))) _Float16 v16h;
typedef __attribute__((ext_vector_type(8)))  float    v8f;

#define Q_CNT 32
#define D_CNT 4096
#define DIM   384
#define H1    384
#define H2    192
#define NPAIR (Q_CNT * D_CNT)

#define NWG      256                 // persistent workgroups
#define WAVES    (NWG * 8)           // 2048 waves
#define NTILES   (NPAIR / 16)        // 8192 16-doc tiles
#define TILE_IT  (NTILES / WAVES)    // 4 tiles per wave

// workspace layout (bytes)
#define WS_DENSE 0
#define WS_A1    (WS_DENSE + Q_CNT * D_CNT * 4)      // 524288
#define WS_DW1   (WS_A1 + Q_CNT * H1 * 4)            // +49152
#define WS_W2P   (WS_DW1 + D_CNT * H1 * 4)           // +6291456  (end ~7.0 MB)

#define W2P_HALFS (12 * 12 * 32 * 16)                // 73728 halves = 147456 B

// ---------------------------------------------------------------------------
// Prep 1: a1[q,k] = b1[k] + sum_j query[q,j] * W1[j,k]   (rows 0..383 of W1)
// ---------------------------------------------------------------------------
__global__ __launch_bounds__(256) void k_qW1(const float* __restrict__ qd,
                                             const float* __restrict__ W1,
                                             const float* __restrict__ b1,
                                             float* __restrict__ a1) {
    int idx = blockIdx.x * 256 + threadIdx.x;
    if (idx >= Q_CNT * H1) return;
    int qi = idx / H1, k = idx % H1;
    float acc = b1[k];
    const float* qr = qd + qi * DIM;
    for (int j = 0; j < DIM; ++j)
        acc += qr[j] * W1[j * H1 + k];          // coalesced across k
    a1[idx] = acc;
}

// ---------------------------------------------------------------------------
// Prep 2: dW1[d,k] = sum_j doc[d,j] * W1[384+j, k]
// ---------------------------------------------------------------------------
__global__ __launch_bounds__(256) void k_dW1(const float* __restrict__ doc,
                                             const float* __restrict__ W1,
                                             float* __restrict__ dW1) {
    int idx = blockIdx.x * 256 + threadIdx.x;    // exactly D_CNT*H1 threads
    int d = idx / H1, k = idx % H1;
    float acc = 0.f;
    const float* dr = doc + d * DIM;
    const float* Wc = W1 + DIM * H1;             // rows 384..767
    for (int j = 0; j < DIM; ++j)
        acc += dr[j] * Wc[j * H1 + k];
    dW1[idx] = acc;
}

// ---------------------------------------------------------------------------
// Prep 3: dense[q,d] = dot(query[q], doc[d])  (query row staged in LDS)
// ---------------------------------------------------------------------------
__global__ __launch_bounds__(256) void k_dense(const float* __restrict__ qd,
                                               const float* __restrict__ doc,
                                               float* __restrict__ dense) {
    __shared__ float qs[DIM];
    int qi = blockIdx.x >> 4;
    int d  = ((blockIdx.x & 15) << 8) + threadIdx.x;
    for (int j = threadIdx.x; j < DIM; j += 256) qs[j] = qd[qi * DIM + j];
    __syncthreads();
    const float4* dr = (const float4*)(doc + d * DIM);
    const float4* qr = (const float4*)qs;
    float acc = 0.f;
    for (int j = 0; j < DIM / 4; ++j) {
        float4 a = qr[j], b = dr[j];
        acc += a.x * b.x + a.y * b.y + a.z * b.z + a.w * b.w;
    }
    dense[qi * D_CNT + d] = acc;
}

// ---------------------------------------------------------------------------
// Prep 4: pack W2 [384x192] f32 -> f16 WMMA B-fragments, kt-major so the
// 12 fragments of one K-step are contiguous (11 KB span -> DS immediates).
// Layout: W2p[((kt*12 + nt)*32 + lane)*16 + j]
//   n = nt*16 + (lane&15);  K = kt*32 + (lane<16 ? 0 : 8) + (j&7) + (j<8 ? 0 : 16)
// ---------------------------------------------------------------------------
__global__ __launch_bounds__(256) void k_packW2(const float* __restrict__ W2,
                                                _Float16* __restrict__ W2p) {
    int idx = blockIdx.x * 256 + threadIdx.x;    // exactly W2P_HALFS threads
    int j    = idx & 15;
    int lane = (idx >> 4) & 31;
    int g    = idx >> 9;                          // kt*12 + nt
    int nt   = g % 12;
    int kt   = g / 12;
    int n = nt * 16 + (lane & 15);
    int k = kt * 32 + ((lane < 16) ? 0 : 8) + (j & 7) + ((j & 8) ? 16 : 0);
    W2p[idx] = (_Float16)W2[k * H2 + n];
}

// ---------------------------------------------------------------------------
// Fused main (persistent): W2 B-fragments staged once per WG into LDS via
// async global->LDS copies; each wave then processes TILE_IT tiles of
// (one query q, 16 docs):
//   h1 built directly in A-fragment registers (rank-2 update, relu, ->f16)
//   h2 = relu(h1 @ W2 + b2) via 144x v_wmma_f32_16x16x32_f16 per tile
//   logit = h2 . w3 + b3 ; out = sigmoid(logit)*dense + (1-sig)*sparse
// ---------------------------------------------------------------------------
__global__ __launch_bounds__(256) void k_fused(
    const float* __restrict__ a1, const float* __restrict__ dW1,
    const float* __restrict__ dense, const float* __restrict__ sparse,
    const float* __restrict__ W1, const _Float16* __restrict__ W2p,
    const float* __restrict__ b2, const float* __restrict__ W3,
    const float* __restrict__ b3, float* __restrict__ out) {

    __shared__ __align__(32) _Float16 sW2[W2P_HALFS];   // 147456 B

    // ---- stage packed W2 into LDS with CDNA5 async global->LDS copies ----
    {
        const char* gbase = (const char*)W2p;
        unsigned tid = threadIdx.x;
#pragma unroll
        for (int it = 0; it < 36; ++it) {               // 36*256*16B = 147456B
            unsigned boff = (it * 256u + tid) * 16u;
            unsigned lds_addr = (unsigned)(uintptr_t)((char*)sW2 + boff);
            unsigned long long gaddr = (unsigned long long)(uintptr_t)(gbase + boff);
            asm volatile("global_load_async_to_lds_b128 %0, %1, off"
                         :: "v"(lds_addr), "v"(gaddr) : "memory");
        }
        asm volatile("s_wait_asynccnt 0x0" ::: "memory");
        __syncthreads();
    }

    const int lane   = threadIdx.x & 31;
    const int waveId = blockIdx.x * 8 + (threadIdx.x >> 5);   // 0..WAVES-1
    const int m  = lane & 15;                  // A-matrix row (doc within tile)
    const int lo = (lane & 16) ? 8 : 0;        // K-chunk select per A layout

    const float b3v = b3[0];

#pragma unroll 1
    for (int it = 0; it < TILE_IT; ++it) {
        const int tile = waveId + it * WAVES;  // 0..NTILES-1, uniform per wave
        const int qi = tile >> 8;
        const int d0 = (tile & 255) << 4;
        const int d  = d0 + m;

        // ---- launder loop-invariant addresses so LICM cannot hoist the
        //      144 B-fragments / W1-row loads out of the tile loop (that
        //      caused 1000+ VGPR live ranges and scratch spills) ----
        unsigned zoff = 0;                          // == 0, but opaque
        const float* w1dv = W1 + 768 * H1;          // dense-score row of W1
        const float* w1sv = W1 + 769 * H1;          // sparse-score row of W1
        asm volatile("" : "+v"(zoff), "+v"(w1dv), "+v"(w1sv));

        const float* a1q = a1 + qi * H1;
        const float* dWd = dW1 + d * H1;
        const float sD = dense[qi * D_CNT + d];
        const float sS = sparse[qi * D_CNT + d];

        v8f acc[12];
#pragma unroll
        for (int nt = 0; nt < 12; ++nt)
#pragma unroll
            for (int r = 0; r < 8; ++r) acc[nt][r] = 0.f;

#pragma unroll
        for (int kt = 0; kt < 12; ++kt) {
            // ---- build A fragment (16 halfs) for this K-step (VALU,
            //      co-executes with the XDL WMMA pipe) ----
            v16h a;
#pragma unroll
            for (int h = 0; h < 2; ++h) {
                int kb = kt * 32 + lo + h * 16;    // 8 contiguous K values
                const float4* A4 = (const float4*)(a1q  + kb);
                const float4* D4 = (const float4*)(dWd  + kb);
                const float4* X4 = (const float4*)(w1dv + kb);
                const float4* S4 = (const float4*)(w1sv + kb);
#pragma unroll
                for (int t = 0; t < 2; ++t) {
                    float4 av = A4[t], dv = D4[t], xv = X4[t], sv = S4[t];
                    int bse = h * 8 + t * 4;
                    a[bse + 0] = (_Float16)fmaxf(av.x + dv.x + sD * xv.x + sS * sv.x, 0.f);
                    a[bse + 1] = (_Float16)fmaxf(av.y + dv.y + sD * xv.y + sS * sv.y, 0.f);
                    a[bse + 2] = (_Float16)fmaxf(av.z + dv.z + sD * xv.z + sS * sv.z, 0.f);
                    a[bse + 3] = (_Float16)fmaxf(av.w + dv.w + sD * xv.w + sS * sv.w, 0.f);
                }
            }
            // ---- 12 N-tiles share this A fragment; B from LDS (kt-major
            //      packing: per-kt fragments contiguous) with two-deep
            //      register prefetch so ds_loads stay 2 WMMAs ahead ----
#define FRAG(KT, NT) ((((KT) * 12 + (NT)) * 32 + lane) << 4)
            v16h bA = *(const v16h*)(sW2 + (FRAG(kt, 0) + zoff));
            v16h bB = *(const v16h*)(sW2 + (FRAG(kt, 1) + zoff));
#pragma unroll
            for (int nt = 0; nt < 12; ++nt) {
                v16h bc = bA;
                bA = bB;
                if (nt < 10)
                    bB = *(const v16h*)(sW2 + (FRAG(kt, nt + 2) + zoff));
                acc[nt] = __builtin_amdgcn_wmma_f32_16x16x32_f16(
                    false, a, false, bc, (short)0, acc[nt], false, false);
            }
#undef FRAG
        }

        // ---- epilogue: relu(+b2), dot with w3
        //      (C layout: VGPR r -> M=r / r+8, lane -> N) ----
        float part[8];
#pragma unroll
        for (int r = 0; r < 8; ++r) part[r] = 0.f;
#pragma unroll
        for (int nt = 0; nt < 12; ++nt) {
            int n = nt * 16 + (lane & 15);
            float bias = b2[n];
            float w3n  = W3[n];
#pragma unroll
            for (int r = 0; r < 8; ++r)
                part[r] += fmaxf(acc[nt][r] + bias, 0.f) * w3n;
        }
        // reduce over the 16 lanes of each half-wave (xor 1,2,4,8 stays in-half)
#pragma unroll
        for (int off = 1; off < 16; off <<= 1)
#pragma unroll
            for (int r = 0; r < 8; ++r)
                part[r] += __shfl_xor(part[r], off, 32);

        if ((lane & 15) == 0) {                // lanes 0 (M=0..7), 16 (M=8..15)
            int mb = (lane < 16) ? 0 : 8;
#pragma unroll
            for (int r = 0; r < 8; ++r) {
                int dd = d0 + mb + r;
                float w = 1.f / (1.f + __expf(-(part[r] + b3v)));
                float dv = dense[qi * D_CNT + dd];
                float sv = sparse[qi * D_CNT + dd];
                out[qi * D_CNT + dd] = w * dv + (1.f - w) * sv;
            }
        }
        if (lane < 16) {                       // outputs 2 & 3: dense / sparse
            out[NPAIR     + qi * D_CNT + d] = sD;
            out[2 * NPAIR + qi * D_CNT + d] = sS;
        }
    }
}

// ---------------------------------------------------------------------------
extern "C" void kernel_launch(void* const* d_in, const int* in_sizes, int n_in,
                              void* d_out, int out_size, void* d_ws, size_t ws_size,
                              hipStream_t stream) {
    const float* qd  = (const float*)d_in[0];   // query_dense [32,384]
    const float* doc = (const float*)d_in[1];   // doc_dense   [4096,384]
    const float* sp  = (const float*)d_in[2];   // sparse      [32,4096]
    const float* W1  = (const float*)d_in[3];   // [770,384]
    const float* b1  = (const float*)d_in[4];   // [384]
    const float* W2  = (const float*)d_in[5];   // [384,192]
    const float* b2  = (const float*)d_in[6];   // [192]
    const float* W3  = (const float*)d_in[7];   // [192,1]
    const float* b3  = (const float*)d_in[8];   // [1]
    float* out = (float*)d_out;                 // [3 * 32*4096] concatenated

    char* ws = (char*)d_ws;
    float*    dense = (float*)(ws + WS_DENSE);
    float*    a1    = (float*)(ws + WS_A1);
    float*    dW1v  = (float*)(ws + WS_DW1);
    _Float16* W2p   = (_Float16*)(ws + WS_W2P);

    k_qW1  <<<(Q_CNT * H1 + 255) / 256, 256, 0, stream>>>(qd, W1, b1, a1);
    k_dW1  <<<(D_CNT * H1) / 256,        256, 0, stream>>>(doc, W1, dW1v);
    k_dense<<<Q_CNT * (D_CNT / 256),     256, 0, stream>>>(qd, doc, dense);
    k_packW2<<<W2P_HALFS / 256,          256, 0, stream>>>(W2, W2p);
    k_fused<<<NWG,                       256, 0, stream>>>(
        a1, dW1v, dense, sp, W1, W2p, b2, W3, b3, out);
}
